// PAM_Module_38233798869168
// MI455X (gfx1250) — matrix operand order repaired and checked by reading, not compiled, not run on hardware
//
#include <hip/hip_runtime.h>

typedef __attribute__((ext_vector_type(16))) __bf16 v16bf;
typedef __attribute__((ext_vector_type(8)))  float  v8f;

#define B_   8
#define C_   512
#define D_   64
#define N_   16384
#define OROWS_ (2 * D_ + C_)      // 640 concatenated Q|K|V rows
#define NSPLIT_ 16
#define LSTRIDE_ 520              // LDS row stride (halves): 1040B, 16B-aligned
#define PARAM_ 10.0f
#define EPS_  1e-6f

#define WMMA_BF16(A, Bf, Cc) \
    __builtin_amdgcn_wmma_f32_16x16x32_bf16(false, (A), false, (Bf), (short)0, (Cc), false, false)

__device__ __forceinline__ float featmap(float t) {
    // param*relu(t) + exp(param*min(t,0)) :  t>0 -> 10t+1 ; else exp(10t)
    return t > 0.0f ? fmaf(PARAM_, t, 1.0f) : __expf(PARAM_ * t);
}

union BfPack8 { uint4 u; __bf16 h[8]; };

// load 8 contiguous bf16 (16B aligned, global or LDS) into elements [e0,e0+8)
__device__ __forceinline__ void load8(v16bf& v, int e0, const __bf16* p) {
    BfPack8 t; t.u = *(const uint4*)p;
#pragma unroll
    for (int e = 0; e < 8; ++e) v[e0 + e] = t.h[e];
}

// ---------------------------------------------------------------------------
// Kernel 0: one-time bf16 conversion of concatenated weights [Q|K|V] rows.
// ---------------------------------------------------------------------------
__global__ void __launch_bounds__(256)
wprep_kernel(const float* __restrict__ Wq, const float* __restrict__ Wk,
             const float* __restrict__ Wv, __bf16* __restrict__ Wc)
{
    int g = blockIdx.x * 256 + threadIdx.x;        // 0 .. 640*512-1
    int o = g >> 9, c = g & (C_ - 1);
    float v;
    if (o < D_)          v = Wq[o * C_ + c];
    else if (o < 2 * D_) v = Wk[(o - D_) * C_ + c];
    else                 v = Wv[(o - 2 * D_) * C_ + c];
    Wc[g] = (__bf16)v;
}

// ---------------------------------------------------------------------------
// Kernel 1: fused Q/K/V projection.  out[o,n] = sum_c Wc[o,c]*x[b,c,n].
// One block owns a 64-n stripe and stages the ENTIRE 512-c column of x into
// LDS (transposed [n][c], stride 520 halves) exactly once; all 5 o-bands
// (640 rows) then loop over the resident stripe -> x read once from HBM.
// Wave = 32o x 32n (2x2 WMMA tiles); 8 waves = 128o x 64n per band.
// Band 0 epilogue is wave-uniform Q or K (feature map); bands 1-4 pure V.
// ---------------------------------------------------------------------------
__global__ void __launch_bounds__(256)
proj_kernel(const float* __restrict__ x, const __bf16* __restrict__ Wc,
            const float* __restrict__ bq, const float* __restrict__ bk,
            const float* __restrict__ bv,
            __bf16* __restrict__ Qf, __bf16* __restrict__ Kf,
            __bf16* __restrict__ Vbf)
{
    extern __shared__ __bf16 xs[];                 // [64][LSTRIDE_] = 65 KB
    const int b     = blockIdx.y;
    const int nbase = blockIdx.x * 64;
    const int lane  = threadIdx.x & 31;
    const int wave  = threadIdx.x >> 5;
    const int wo    = wave >> 1;                   // 0..3 : 32-row o band
    const int wn    = wave & 1;                    // 0..1 : 32-col n band
    const int r     = lane & 15;
    const int kb    = (lane < 16) ? 0 : 8;
    const int kbB   = (lane < 16) ? 0 : 8;
    const int hi8   = (lane < 16) ? 0 : 8;

    // ---- stage x[b, 0:512, nbase:nbase+64] -> LDS bf16 transposed --------
    const size_t xbase = (size_t)b * C_ * N_ + nbase;
#pragma unroll 4
    for (int i = 0; i < 32; ++i) {                 // 8192 float4 / 256 thr
        int idx = i * 256 + threadIdx.x;
        int n4 = (idx & 15) * 4, cc = idx >> 4;    // coalesced 16B per lane
        float4 v = *(const float4*)&x[xbase + (size_t)cc * N_ + n4];
        xs[(n4 + 0) * LSTRIDE_ + cc] = (__bf16)v.x;
        xs[(n4 + 1) * LSTRIDE_ + cc] = (__bf16)v.y;
        xs[(n4 + 2) * LSTRIDE_ + cc] = (__bf16)v.z;
        xs[(n4 + 3) * LSTRIDE_ + cc] = (__bf16)v.w;
    }
    __syncthreads();

    const __bf16* xrow0 = xs + (size_t)(wn * 32 + r) * LSTRIDE_;
    const __bf16* xrow1 = xrow0 + (size_t)16 * LSTRIDE_;
    const int n0 = nbase + wn * 32 + r;

    // ---- loop the 5 o-bands over the resident stripe ---------------------
#pragma unroll 1
    for (int ob = 0; ob < OROWS_ / 128; ++ob) {
        const int obase = ob * 128;
        const __bf16* ar0 = Wc + (size_t)(obase + wo * 32 + r) * C_;
        const __bf16* ar1 = ar0 + (size_t)16 * C_;

        v8f a00 = {}, a01 = {}, a10 = {}, a11 = {};
#pragma unroll 2
        for (int c0 = 0; c0 < C_; c0 += 32) {
            v16bf fa0, fa1, fb0, fb1;
            load8(fa0, 0, ar0 + c0 + kb);    load8(fa0, 8, ar0 + c0 + kb + 16);
            load8(fa1, 0, ar1 + c0 + kb);    load8(fa1, 8, ar1 + c0 + kb + 16);
            load8(fb0, 0, xrow0 + c0 + kbB); load8(fb0, 8, xrow0 + c0 + kbB + 16);
            load8(fb1, 0, xrow1 + c0 + kbB); load8(fb1, 8, xrow1 + c0 + kbB + 16);
            a00 = WMMA_BF16(fa0, fb0, a00);
            a01 = WMMA_BF16(fa0, fb1, a01);
            a10 = WMMA_BF16(fa1, fb0, a10);
            a11 = WMMA_BF16(fa1, fb1, a11);
        }

        const int ot0 = obase + wo * 32;           // row base of tile pair
        if (ob == 0) {
            if (wo < 2) {                          // rows 0..63 : all Q
                auto epiQ = [&](const v8f& acc, int ot, int n) {
#pragma unroll
                    for (int j = 0; j < 8; ++j) {
                        int o = ot + j + hi8;
                        float v = featmap(acc[j] + bq[o]);
                        Qf[((size_t)b * N_ + n) * D_ + o] = (__bf16)v;
                    }
                };
                epiQ(a00, ot0,      n0);
                epiQ(a01, ot0,      n0 + 16);
                epiQ(a10, ot0 + 16, n0);
                epiQ(a11, ot0 + 16, n0 + 16);
            } else {                               // rows 64..127 : all K
                auto epiK = [&](const v8f& acc, int ot, int n) {
#pragma unroll
                    for (int j = 0; j < 8; ++j) {
                        int m = ot - D_ + j + hi8;
                        float v = featmap(acc[j] + bk[m]);
                        Kf[((size_t)b * D_ + m) * N_ + n] = (__bf16)v;
                    }
                };
                epiK(a00, ot0,      n0);
                epiK(a01, ot0,      n0 + 16);
                epiK(a10, ot0 + 16, n0);
                epiK(a11, ot0 + 16, n0 + 16);
            }
        } else {                                   // bands 1..4 : all V
            auto epiV = [&](const v8f& acc, int ot, int n) {
#pragma unroll
                for (int j = 0; j < 8; ++j) {
                    int c = ot - 2 * D_ + j + hi8;
                    float v = acc[j] + bv[c];
                    Vbf[((size_t)b * C_ + c) * N_ + n] = (__bf16)v;
                }
            };
            epiV(a00, ot0,      n0);
            epiV(a01, ot0,      n0 + 16);
            epiV(a10, ot0 + 16, n0);
            epiV(a11, ot0 + 16, n0 + 16);
        }
    }
}

// ---------------------------------------------------------------------------
// Kernel 2: KV[b,m,c] = sum_n K[b,m,n]*V[b,c,n].  Wave = 32m x 32c (2x2),
// block = 2 m-bands x 4 c-bands = 64m x 128c.  N split NSPLIT_ ways into
// deterministic partial buffers.  All fragments contiguous uint4 loads.
// ---------------------------------------------------------------------------
__global__ void __launch_bounds__(256)
kv_kernel(const __bf16* __restrict__ Kf, const __bf16* __restrict__ Vbf,
          float* __restrict__ KVp)
{
    const int b      = blockIdx.z;
    const int cbase  = blockIdx.x * 128;
    const int nsplit = blockIdx.y;
    const int nstart = nsplit * (N_ / NSPLIT_);
    const int nend   = nstart + (N_ / NSPLIT_);
    const int lane = threadIdx.x & 31;
    const int wave = threadIdx.x >> 5;
    const int wm = wave >> 2;                      // 0..1 : 32-row m band
    const int wc = wave & 3;                       // 0..3 : 32-col c band
    const int r  = lane & 15;
    const int kb  = (lane < 16) ? 0 : 8;
    const int kbB = (lane < 16) ? 0 : 8;
    const int hi8 = (lane < 16) ? 0 : 8;

    const __bf16* arow0 = Kf  + ((size_t)b * D_ + wm * 32 + r) * N_;
    const __bf16* arow1 = arow0 + (size_t)16 * N_;
    const __bf16* brow0 = Vbf + ((size_t)b * C_ + cbase + wc * 32 + r) * N_;
    const __bf16* brow1 = brow0 + (size_t)16 * N_;

    v8f a00 = {}, a01 = {}, a10 = {}, a11 = {};
#pragma unroll 2
    for (int n0 = nstart; n0 < nend; n0 += 32) {
        if (n0 + 128 < nend) {                     // stream-ahead hints
            __builtin_prefetch(arow0 + n0 + 128, 0, 1);
            __builtin_prefetch(brow0 + n0 + 128, 0, 1);
            __builtin_prefetch(brow1 + n0 + 128, 0, 1);
        }
        v16bf fa0, fa1, fb0, fb1;
        load8(fa0, 0, arow0 + n0 + kb);   load8(fa0, 8, arow0 + n0 + kb + 16);
        load8(fa1, 0, arow1 + n0 + kb);   load8(fa1, 8, arow1 + n0 + kb + 16);
        load8(fb0, 0, brow0 + n0 + kbB);  load8(fb0, 8, brow0 + n0 + kbB + 16);
        load8(fb1, 0, brow1 + n0 + kbB);  load8(fb1, 8, brow1 + n0 + kbB + 16);
        a00 = WMMA_BF16(fa0, fb0, a00);
        a01 = WMMA_BF16(fa0, fb1, a01);
        a10 = WMMA_BF16(fa1, fb0, a10);
        a11 = WMMA_BF16(fa1, fb1, a11);
    }

    float* dst = KVp + (size_t)nsplit * (B_ * D_ * C_) + (size_t)b * D_ * C_;
    auto epi = [&](const v8f& acc, int mt, int ct) {
#pragma unroll
        for (int j = 0; j < 8; ++j)
            dst[(size_t)(mt + j + hi8) * C_ + ct + r] = acc[j];
    };
    const int mb = wm * 32, cb = cbase + wc * 32;
    epi(a00, mb,      cb);
    epi(a01, mb,      cb + 16);
    epi(a10, mb + 16, cb);
    epi(a11, mb + 16, cb + 16);
}

// Kernel 3: reduce the NSPLIT_ partials; emit bf16 KV transposed [b,c,m]
// so the output GEMM's A fragments are contiguous.
__global__ void __launch_bounds__(256)
kvred_kernel(const float* __restrict__ KVp, __bf16* __restrict__ KVt)
{
    int g = blockIdx.x * 256 + threadIdx.x;        // 0 .. B*D*C-1
    float s = 0.f;
#pragma unroll
    for (int i = 0; i < NSPLIT_; ++i) s += KVp[(size_t)i * (B_ * D_ * C_) + g];
    int c = g & (C_ - 1);
    int bm = g >> 9;
    int m = bm & (D_ - 1);
    int b = bm >> 6;
    KVt[((size_t)b * C_ + c) * D_ + m] = (__bf16)s;
}

// Kernel 4: Ksum[b,m] = sum_n K[b,m,n]  (vector loads + LDS tree reduction)
__global__ void __launch_bounds__(256)
ksum_kernel(const __bf16* __restrict__ Kf, float* __restrict__ Ksum)
{
    const int bm = blockIdx.x;                     // 0 .. B*D-1
    const __bf16* row = Kf + (size_t)bm * N_;
    float s = 0.f;
    for (int i = threadIdx.x; i < N_ / 8; i += 256) {
        BfPack8 t; t.u = *(const uint4*)(row + i * 8);
#pragma unroll
        for (int e = 0; e < 8; ++e) s += (float)t.h[e];
    }
    __shared__ float red[256];
    red[threadIdx.x] = s; __syncthreads();
    for (int st = 128; st > 0; st >>= 1) {
        if (threadIdx.x < st) red[threadIdx.x] += red[threadIdx.x + st];
        __syncthreads();
    }
    if (threadIdx.x == 0) Ksum[bm] = red[0];
}

// Kernel 5: norm[b,n] = 1 / sum_m Q[b,n,m] * (Ksum[b,m] + eps)
__global__ void __launch_bounds__(256)
norm_kernel(const __bf16* __restrict__ Qf, const float* __restrict__ Ksum,
            float* __restrict__ nrm)
{
    int g = blockIdx.x * 256 + threadIdx.x;        // 0 .. B*N-1
    int b = g >> 14;                               // N_ = 2^14
    const __bf16* q = Qf + (size_t)g * D_;
    const float* s  = Ksum + b * D_;
    float dot = 0.f;
#pragma unroll
    for (int i = 0; i < 8; ++i) {
        BfPack8 t; t.u = *(const uint4*)(q + i * 8);
#pragma unroll
        for (int e = 0; e < 8; ++e) dot += (float)t.h[e] * (s[i * 8 + e] + EPS_);
    }
    nrm[g] = 1.0f / dot;
}

// ---------------------------------------------------------------------------
// Kernel 6: out[b,c,n] = x + gamma*norm[n]*sum_m KVt[b,c,m]*Q[b,n,m].
// Wave = 32c x 32n (2x2); block = 2 c-bands x 4 n-bands.  The wave's 4 Q
// B-fragments (K=64) are preloaded ONCE; the c dimension loops inside the
// kernel (KVt is 512KB, L2-resident) -> Qf read once from HBM.
// Stores are contiguous 16-dword runs per half-wave.
// ---------------------------------------------------------------------------
__global__ void __launch_bounds__(256)
out_kernel(const float* __restrict__ x, const __bf16* __restrict__ Qf,
           const __bf16* __restrict__ KVt, const float* __restrict__ nrm,
           const float* __restrict__ gam, float* __restrict__ out)
{
    const int b     = blockIdx.y;
    const int nbase = blockIdx.x * 128;
    const int lane = threadIdx.x & 31;
    const int wave = threadIdx.x >> 5;
    const int wc = wave & 1;                       // 0..1 : 32-row c band
    const int wn = wave >> 1;                      // 0..3 : 32-col n band
    const int r  = lane & 15;
    const int kb  = (lane < 16) ? 0 : 8;
    const int kbB = (lane < 16) ? 0 : 8;
    const int hi8 = (lane < 16) ? 0 : 8;

    const int n0 = nbase + wn * 32 + r;
    const __bf16* brow0 = Qf + ((size_t)b * N_ + n0) * D_;
    const __bf16* brow1 = brow0 + (size_t)16 * D_;

    // preload all Q B-fragments: [kstep][n-tile]
    v16bf qb[2][2];
#pragma unroll
    for (int ks = 0; ks < 2; ++ks) {
        const int m0 = ks * 32;
        load8(qb[ks][0], 0, brow0 + m0 + kbB); load8(qb[ks][0], 8, brow0 + m0 + kbB + 16);
        load8(qb[ks][1], 0, brow1 + m0 + kbB); load8(qb[ks][1], 8, brow1 + m0 + kbB + 16);
    }

    const float g   = gam[0];
    const float nr0 = nrm[(size_t)b * N_ + n0];
    const float nr1 = nrm[(size_t)b * N_ + n0 + 16];

#pragma unroll 1
    for (int cb = 0; cb < C_; cb += 64) {
        const __bf16* ar0 = KVt + ((size_t)b * C_ + cb + wc * 32 + r) * D_;
        const __bf16* ar1 = ar0 + (size_t)16 * D_;

        v8f a00 = {}, a01 = {}, a10 = {}, a11 = {};
#pragma unroll
        for (int ks = 0; ks < 2; ++ks) {
            const int m0 = ks * 32;
            v16bf fa0, fa1;
            load8(fa0, 0, ar0 + m0 + kb);  load8(fa0, 8, ar0 + m0 + kb + 16);
            load8(fa1, 0, ar1 + m0 + kb);  load8(fa1, 8, ar1 + m0 + kb + 16);
            a00 = WMMA_BF16(fa0, qb[ks][0], a00);
            a01 = WMMA_BF16(fa0, qb[ks][1], a01);
            a10 = WMMA_BF16(fa1, qb[ks][0], a10);
            a11 = WMMA_BF16(fa1, qb[ks][1], a11);
        }

        auto epi = [&](const v8f& acc, int ct, int n, float nr) {
#pragma unroll
            for (int j = 0; j < 8; ++j) {
                size_t idx = ((size_t)b * C_ + ct + j + hi8) * (size_t)N_ + n;
                out[idx] = x[idx] + g * nr * acc[j];
            }
        };
        const int c0 = cb + wc * 32;
        epi(a00, c0,      n0,      nr0);
        epi(a01, c0,      n0 + 16, nr1);
        epi(a10, c0 + 16, n0,      nr0);
        epi(a11, c0 + 16, n0 + 16, nr1);
    }
}

extern "C" void kernel_launch(void* const* d_in, const int* in_sizes, int n_in,
                              void* d_out, int out_size, void* d_ws, size_t ws_size,
                              hipStream_t stream) {
    (void)in_sizes; (void)n_in; (void)out_size; (void)ws_size;
    const float* x  = (const float*)d_in[0];
    const float* Wq = (const float*)d_in[1];
    const float* bq = (const float*)d_in[2];
    const float* Wk = (const float*)d_in[3];
    const float* bk = (const float*)d_in[4];
    const float* Wv = (const float*)d_in[5];
    const float* bv = (const float*)d_in[6];
    const float* gm = (const float*)d_in[7];
    float* out = (float*)d_out;

    char* w = (char*)d_ws;
    __bf16* Qf  = (__bf16*)w; w += (size_t)B_ * N_ * D_ * 2;          // 16 MB
    __bf16* Kf  = (__bf16*)w; w += (size_t)B_ * D_ * N_ * 2;          // 16 MB
    __bf16* Vb  = (__bf16*)w; w += (size_t)B_ * C_ * N_ * 2;          // 128 MB
    __bf16* Wc  = (__bf16*)w; w += (size_t)OROWS_ * C_ * 2;           // 640 KB
    float*  KVp = (float*)w;  w += (size_t)NSPLIT_ * B_ * D_ * C_ * 4;// 16 MB
    __bf16* KVt = (__bf16*)w; w += (size_t)B_ * C_ * D_ * 2;          // 512 KB
    float*  Ks  = (float*)w;  w += (size_t)B_ * D_ * 4;
    float*  Nr  = (float*)w;                                          // 512 KB

    const size_t lds_bytes = (size_t)64 * LSTRIDE_ * 2;               // 65 KB

    wprep_kernel<<<(OROWS_ * C_) / 256, 256, 0, stream>>>(Wq, Wk, Wv, Wc);
    proj_kernel<<<dim3(N_ / 64, B_), 256, lds_bytes, stream>>>(
        x, Wc, bq, bk, bv, Qf, Kf, Vb);
    kv_kernel<<<dim3(C_ / 128, NSPLIT_, B_), 256, 0, stream>>>(Kf, Vb, KVp);
    kvred_kernel<<<(B_ * D_ * C_) / 256, 256, 0, stream>>>(KVp, KVt);
    ksum_kernel<<<B_ * D_, 256, 0, stream>>>(Kf, Ks);
    norm_kernel<<<(B_ * N_) / 256, 256, 0, stream>>>(Qf, Ks, Nr);
    out_kernel<<<dim3(N_ / 128, B_), 256, 0, stream>>>(
        x, Qf, KVt, Nr, gm, out);
}
